// GATConv_4647154614657
// MI455X (gfx1250) — matrix operand order repaired and checked by reading, not compiled, not run on hardware
//
#include <hip/hip_runtime.h>
#include <hip/hip_bf16.h>
#include <cstdint>

typedef _Float16 half_t;
typedef __attribute__((ext_vector_type(16))) _Float16 v16h;
typedef __attribute__((ext_vector_type(8)))  float    v8f;

#define BATCH 8
#define NN    2048
#define FF    128

// =============================================================================
// Kernel 1: h = x @ W^T (f16 WMMA), write hT[b][f][n] (f16, transposed),
//           and s = h.a_src, d = h.a_dst via cross-lane reduction.
// One wave per 16-row M-tile. A/B packed per CDNA5 ISA 16-bit WMMA layouts:
//   A (16x32): lane M = lane&15; element e -> K = (e<8 ? half*8+e : 16+half*8+e-8)
//   B (32x16): lane N = lane&15; element e -> K = (e<8 ? half*8+e : 16+half*8+e-8)   (half = lane>>4)
//   C (16x16): vgpr r, lane&15 = N, M = r + 8*half
// =============================================================================
__global__ __launch_bounds__(128) void gat_h_kernel(
    const float* __restrict__ x, const float* __restrict__ W,
    const float* __restrict__ a, half_t* __restrict__ hT,
    float* __restrict__ Sp, float* __restrict__ Dp)
{
    const int lane = threadIdx.x & 31;
    const int wave = threadIdx.x >> 5;
    const int m0   = (blockIdx.x * 4 + wave) * 16;
    const int r16  = lane & 15;
    const int halfg = lane >> 4;
    const int row  = m0 + r16;

    // ---- load & pack A tiles for all 4 K-chunks (K = 128 total) ----
    v16h A[4];
#pragma unroll
    for (int kc = 0; kc < 4; ++kc) {
        const float* xr = x + row * FF + kc * 32 + halfg * 8;
        float4 f0 = *(const float4*)(xr);
        float4 f1 = *(const float4*)(xr + 4);
        float4 f2 = *(const float4*)(xr + 16);
        float4 f3 = *(const float4*)(xr + 20);
        v16h t;
        t[0]=(half_t)f0.x; t[1]=(half_t)f0.y; t[2]=(half_t)f0.z; t[3]=(half_t)f0.w;
        t[4]=(half_t)f1.x; t[5]=(half_t)f1.y; t[6]=(half_t)f1.z; t[7]=(half_t)f1.w;
        t[8]=(half_t)f2.x; t[9]=(half_t)f2.y; t[10]=(half_t)f2.z; t[11]=(half_t)f2.w;
        t[12]=(half_t)f3.x; t[13]=(half_t)f3.y; t[14]=(half_t)f3.z; t[15]=(half_t)f3.w;
        A[kc] = t;
    }

    float sacc[8], dacc[8];
#pragma unroll
    for (int r = 0; r < 8; ++r) { sacc[r] = 0.f; dacc[r] = 0.f; }

#pragma unroll
    for (int ft = 0; ft < 8; ++ft) {
        const int col = ft * 16 + r16;        // output feature this lane owns (B: N index)
        v8f c;
#pragma unroll
        for (int r = 0; r < 8; ++r) c[r] = 0.f;
#pragma unroll
        for (int kc = 0; kc < 4; ++kc) {
            const float* wr = W + col * FF + kc * 32 + halfg * 8;  // W[col][k] contiguous
            float4 g0 = *(const float4*)(wr);
            float4 g1 = *(const float4*)(wr + 4);
            float4 g2 = *(const float4*)(wr + 16);
            float4 g3 = *(const float4*)(wr + 20);
            v16h bt;
            bt[0]=(half_t)g0.x; bt[1]=(half_t)g0.y; bt[2]=(half_t)g0.z; bt[3]=(half_t)g0.w;
            bt[4]=(half_t)g1.x; bt[5]=(half_t)g1.y; bt[6]=(half_t)g1.z; bt[7]=(half_t)g1.w;
            bt[8]=(half_t)g2.x; bt[9]=(half_t)g2.y; bt[10]=(half_t)g2.z; bt[11]=(half_t)g2.w;
            bt[12]=(half_t)g3.x; bt[13]=(half_t)g3.y; bt[14]=(half_t)g3.z; bt[15]=(half_t)g3.w;
            c = __builtin_amdgcn_wmma_f32_16x16x32_f16(
                    false, A[kc], false, bt, (short)0, c, false, false);
        }
        const float as = a[col];        // a_src
        const float ad = a[FF + col];   // a_dst
#pragma unroll
        for (int r = 0; r < 8; ++r) {
            const int g  = m0 + r + 8 * halfg;     // global row of this C element
            const int bb = g >> 11;                // batch
            const int nn = g & (NN - 1);           // node
            hT[((size_t)(bb * FF + col) << 11) + nn] = (half_t)c[r];
            sacc[r] += c[r] * as;
            dacc[r] += c[r] * ad;
        }
    }

    // reduce s/d across the 16 lanes (N dimension) of each half
#pragma unroll
    for (int r = 0; r < 8; ++r) {
        float s = sacc[r], dd = dacc[r];
#pragma unroll
        for (int off = 8; off; off >>= 1) {
            s  += __shfl_xor(s,  off, 16);
            dd += __shfl_xor(dd, off, 16);
        }
        if (r16 == 0) {
            const int g = m0 + r + 8 * halfg;
            Sp[g] = s;
            Dp[g] = dd;
        }
    }
}

// =============================================================================
// Kernel 2: per (b, i): m_i = max_j e_ij, l_i = sum_j exp(e_ij - m_i),
//           e_ij = mask(adj, lrelu(s_i + d_j)). One block per row i; adj row
//           cached in registers across the batch loop. One-pass online softmax.
// =============================================================================
__global__ __launch_bounds__(256) void gat_ml_kernel(
    const float* __restrict__ Sp, const float* __restrict__ Dp,
    const int* __restrict__ adj, float* __restrict__ Mp, float* __restrict__ Lp)
{
    const int i   = blockIdx.x;
    const int tid = threadIdx.x;
    const int* arow = adj + (size_t)i * NN;
    const int j0 = tid * 8;
    const int4 q0 = *(const int4*)(arow + j0);
    const int4 q1 = *(const int4*)(arow + j0 + 4);
    int av[8] = {q0.x, q0.y, q0.z, q0.w, q1.x, q1.y, q1.z, q1.w};

    __shared__ float sm[8], sl[8];

    for (int b = 0; b < BATCH; ++b) {
        const float si = Sp[b * NN + i];
        const float* drow = Dp + b * NN;
        const float4 d0 = *(const float4*)(drow + j0);
        const float4 d1 = *(const float4*)(drow + j0 + 4);
        float dv[8] = {d0.x, d0.y, d0.z, d0.w, d1.x, d1.y, d1.z, d1.w};

        float m = -__builtin_inff(), l = 0.f;
#pragma unroll
        for (int t = 0; t < 8; ++t) {
            float e = si + dv[t];
            e = e > 0.f ? e : 0.2f * e;
            e = (av[t] == 0) ? -1.0e9f : e;
            const float mn = fmaxf(m, e);
            l = l * __expf(m - mn) + __expf(e - mn);
            m = mn;
        }
        // wave32 reduce
#pragma unroll
        for (int off = 16; off; off >>= 1) {
            const float mo = __shfl_xor(m, off, 32);
            const float lo = __shfl_xor(l, off, 32);
            const float mn = fmaxf(m, mo);
            l = l * __expf(m - mn) + lo * __expf(mo - mn);
            m = mn;
        }
        if ((tid & 31) == 0) { sm[tid >> 5] = m; sl[tid >> 5] = l; }
        __syncthreads();
        if (tid == 0) {
            float mr = sm[0], lr = sl[0];
#pragma unroll
            for (int w = 1; w < 8; ++w) {
                const float mn = fmaxf(mr, sm[w]);
                lr = lr * __expf(mr - mn) + sl[w] * __expf(sm[w] - mn);
                mr = mn;
            }
            Mp[b * NN + i] = mr;
            Lp[b * NN + i] = lr;
        }
        __syncthreads();
    }
}

// =============================================================================
// Kernel 3: out[b,i,:] = (1/l_i) * sum_j exp(e_ij - m_i) * h[b,j,:]
// Block = 4 waves = 64 rows i; stream j in K=32 chunks; hT chunk (128f x 32j,
// 8 KB f16) double-buffered through LDS via GLOBAL_LOAD_ASYNC_TO_LDS_B128
// (ASYNCcnt-tracked, overlaps with WMMA on the other buffer); P tile built
// on the fly in A-matrix lane layout; 8 f16 WMMAs per chunk per wave.
// =============================================================================
__global__ __launch_bounds__(128) void gat_attn_kernel(
    const half_t* __restrict__ hT, const int* __restrict__ adj,
    const float* __restrict__ Sp, const float* __restrict__ Dp,
    const float* __restrict__ Mp, const float* __restrict__ Lp,
    float* __restrict__ out)
{
    __shared__ _Float16 lds[2][FF * 32];

    const int b    = blockIdx.y;
    const int i0   = blockIdx.x * 64;
    const int tid  = threadIdx.x;
    const int lane = tid & 31;
    const int wave = tid >> 5;
    const int r16  = lane & 15;
    const int halfg = lane >> 4;
    const int i0w  = i0 + wave * 16;
    const int irow = i0w + r16;

    const float si = Sp[b * NN + irow];
    const float mi = Mp[b * NN + irow];
    const half_t* hTb = hT + ((size_t)b << 18);    // b * 128 * 2048
    const int* adjrow = adj + (size_t)irow * NN;
    const float* drow = Dp + b * NN;

    v8f acc[8];
#pragma unroll
    for (int ft = 0; ft < 8; ++ft)
#pragma unroll
        for (int r = 0; r < 8; ++r) acc[ft][r] = 0.f;

    // Async stage: thread t copies hT row f=t, 32 j-values (64 B) straight
    // into LDS[f][0..31] with 4x global_load_async_to_lds_b128 (no VGPR
    // round-trip; tracked by ASYNCcnt).
    auto stage_async = [&](int chunk, int buf) {
        const half_t* gsrc = hTb + ((size_t)tid << 11) + chunk * 32;
        _Float16* ldst = &lds[buf][tid * 32];
#pragma unroll
        for (int p = 0; p < 4; ++p) {
            const uint32_t laddr = (uint32_t)(uintptr_t)(ldst + p * 8); // addr[31:0] = LDS addr
            asm volatile("global_load_async_to_lds_b128 %0, %1, off"
                         :
                         : "v"(laddr), "v"(gsrc + p * 8)
                         : "memory");
        }
    };

    stage_async(0, 0);

    for (int c = 0; c < 64; ++c) {
        // Only this iteration's buffer has outstanding async ops here.
        asm volatile("s_wait_asynccnt 0" ::: "memory");
        __syncthreads();                       // cross-wave visibility; buf c^1 free
        if (c + 1 < 64) stage_async(c + 1, (c + 1) & 1);
        if (c + 2 < 64)                        // warm L2/L0 for chunk after next
            __builtin_prefetch(adjrow + (c + 2) * 32, 0, 1);

        // ---- build P tile (A matrix, 16x32 f16) for this j-chunk ----
        const int jbase = c * 32;
        const int jA = jbase + halfg * 16;
        const int jB = jbase + halfg * 16 + 8;
        const int4 q0 = *(const int4*)(adjrow + jA);
        const int4 q1 = *(const int4*)(adjrow + jA + 4);
        const int4 q2 = *(const int4*)(adjrow + jB);
        const int4 q3 = *(const int4*)(adjrow + jB + 4);
        const float4 d0 = *(const float4*)(drow + jA);
        const float4 d1 = *(const float4*)(drow + jA + 4);
        const float4 d2 = *(const float4*)(drow + jB);
        const float4 d3 = *(const float4*)(drow + jB + 4);

        auto pmake = [&](int av, float dv) -> _Float16 {
            float e = si + dv;
            e = e > 0.f ? e : 0.2f * e;
            e = (av == 0) ? -1.0e9f : e;
            return (_Float16)__expf(e - mi);
        };
        v16h P;
        P[0]=pmake(q0.x,d0.x); P[1]=pmake(q0.y,d0.y); P[2]=pmake(q0.z,d0.z); P[3]=pmake(q0.w,d0.w);
        P[4]=pmake(q1.x,d1.x); P[5]=pmake(q1.y,d1.y); P[6]=pmake(q1.z,d1.z); P[7]=pmake(q1.w,d1.w);
        P[8]=pmake(q2.x,d2.x); P[9]=pmake(q2.y,d2.y); P[10]=pmake(q2.z,d2.z); P[11]=pmake(q2.w,d2.w);
        P[12]=pmake(q3.x,d3.x); P[13]=pmake(q3.y,d3.y); P[14]=pmake(q3.z,d3.z); P[15]=pmake(q3.w,d3.w);

        // ---- 8 WMMAs: acc[ft] += P @ hT_chunk[:, ft*16 + r16] ----
        const _Float16* lbase = &lds[c & 1][0];
#pragma unroll
        for (int ft = 0; ft < 8; ++ft) {
            const v16h* bp = (const v16h*)(lbase + (ft * 16 + r16) * 32 + halfg * 16);
            acc[ft] = __builtin_amdgcn_wmma_f32_16x16x32_f16(
                          false, P, false, *bp, (short)0, acc[ft], false, false);
        }
    }

    // ---- epilogue: divide by l_i, store f32 output ----
#pragma unroll
    for (int r = 0; r < 8; ++r) {
        const int gi = i0w + r + 8 * halfg;
        const float linv = 1.f / Lp[b * NN + gi];
#pragma unroll
        for (int ft = 0; ft < 8; ++ft) {
            out[((size_t)b * NN + gi) * FF + ft * 16 + r16] = acc[ft][r] * linv;
        }
    }
}

// =============================================================================
extern "C" void kernel_launch(void* const* d_in, const int* in_sizes, int n_in,
                              void* d_out, int out_size, void* d_ws, size_t ws_size,
                              hipStream_t stream) {
    const float* x   = (const float*)d_in[0];   // [8,2048,128] f32
    const int*   adj = (const int*)d_in[1];     // [2048,2048] i32
    const float* W   = (const float*)d_in[2];   // [128,128] f32
    const float* a   = (const float*)d_in[3];   // [256,1] f32
    float* out = (float*)d_out;                 // [8,2048,128] f32

    char* ws = (char*)d_ws;
    half_t* hT = (half_t*)ws;                          // 8*128*2048*2 = 4 MB
    float*  Sp = (float*)(ws + (4u << 20));            // 16384 f32
    float*  Dp = Sp + BATCH * NN;
    float*  Mp = Dp + BATCH * NN;
    float*  Lp = Mp + BATCH * NN;

    gat_h_kernel<<<dim3(256), dim3(128), 0, stream>>>(x, W, a, hT, Sp, Dp);
    gat_ml_kernel<<<dim3(NN), dim3(256), 0, stream>>>(Sp, Dp, adj, Mp, Lp);
    gat_attn_kernel<<<dim3(32, BATCH), dim3(128), 0, stream>>>(hT, adj, Sp, Dp, Mp, Lp, out);
}
